// EncoderorthoRNN_2525440770632
// MI455X (gfx1250) — compile-verified, matrix operand order checked by reading, not run on hardware
//
#include <hip/hip_runtime.h>
#include <hip/hip_bf16.h>

// ---------------------------------------------------------------------------
// LSTM encoder, MI455X (gfx1250, wave32, WMMA).
//   B=64, T=512, E=300, HC=512 (gates 4*HC=2048)
// Phase 1: fused embedding-gather + input GEMM -> xg[T][B][2048] (f32, ws)
// Phase 2: persistent recurrence kernel, 32 blocks x 128 thr (128 waves),
//          W_hh cached in LDS (bf16, WMMA B layout), one grid barrier/step.
// ---------------------------------------------------------------------------

#define BB   64
#define TT   512
#define EE   300
#define HCC  512
#define G4   2048      // 4*HC
#define NBLK 32        // persistent blocks in recurrence kernel

typedef __attribute__((ext_vector_type(16))) __bf16 v16bf;
typedef __attribute__((ext_vector_type(8)))  __bf16 v8bf;
typedef __attribute__((ext_vector_type(8)))  float  v8f;

union V16 { v16bf v; __bf16 e[16]; };

__device__ __forceinline__ float sigmoidf_(float x) {
    return 1.0f / (1.0f + __expf(-x));
}

// Load 16 bf16 A/B fragment elements for one wave lane from an f32 row:
// K = k0+khalf+0..7 and k0+16+khalf+0..7  (matches 16-bit 16x32 WMMA layout).
__device__ __forceinline__ v16bf load_frag_f32(const float* __restrict__ row,
                                               int k0, int khalf, bool tail) {
    V16 x;
    if (!tail) {
        const float4 f0 = *(const float4*)(row + k0 + khalf);
        const float4 f1 = *(const float4*)(row + k0 + khalf + 4);
        const float4 f2 = *(const float4*)(row + k0 + 16 + khalf);
        const float4 f3 = *(const float4*)(row + k0 + 16 + khalf + 4);
        x.e[0]=(__bf16)f0.x; x.e[1]=(__bf16)f0.y; x.e[2]=(__bf16)f0.z; x.e[3]=(__bf16)f0.w;
        x.e[4]=(__bf16)f1.x; x.e[5]=(__bf16)f1.y; x.e[6]=(__bf16)f1.z; x.e[7]=(__bf16)f1.w;
        x.e[8]=(__bf16)f2.x; x.e[9]=(__bf16)f2.y; x.e[10]=(__bf16)f2.z; x.e[11]=(__bf16)f2.w;
        x.e[12]=(__bf16)f3.x; x.e[13]=(__bf16)f3.y; x.e[14]=(__bf16)f3.z; x.e[15]=(__bf16)f3.w;
    } else {
        #pragma unroll
        for (int j = 0; j < 8; ++j) {
            int k = k0 + khalf + j;
            x.e[j] = (k < EE) ? (__bf16)row[k] : (__bf16)0.0f;
        }
        #pragma unroll
        for (int j = 0; j < 8; ++j) {
            int k = k0 + 16 + khalf + j;
            x.e[8 + j] = (k < EE) ? (__bf16)row[k] : (__bf16)0.0f;
        }
    }
    return x.v;
}

// ---------------------------------------------------------------------------
// Kernel 0: zero the h double-buffer (bf16) and the grid-barrier counter.
// ---------------------------------------------------------------------------
__global__ void k_init(__bf16* __restrict__ hbuf, unsigned* __restrict__ cnt) {
    int i = blockIdx.x * blockDim.x + threadIdx.x;
    if (i < 2 * BB * HCC) hbuf[i] = (__bf16)0.0f;
    if (i == 0) *cnt = 0u;
}

// ---------------------------------------------------------------------------
// Kernel 1: xg[t*64+b][g] = emb[seq[b][t]] . W_ih[g] + b_ih[g] + b_hh[g]
// One 16x16 tile per wave; K padded 300 -> 320 (10 k-iters of 32, bf16 WMMA).
// grid = 32768 blocks x 256 thr (8 waves) -> 262144 tiles = 2048 x 128.
// ---------------------------------------------------------------------------
__global__ void __launch_bounds__(256)
k_embed_gemm(const int* __restrict__ seq, const float* __restrict__ emb,
             const float* __restrict__ W_ih, const float* __restrict__ b_ih,
             const float* __restrict__ b_hh, float* __restrict__ xg) {
    const int lane  = threadIdx.x & 31;
    const int wid   = threadIdx.x >> 5;
    const int tile  = blockIdx.x * 8 + wid;     // 0 .. 262143
    const int rt    = tile >> 7;                // row tile (B*T/16)
    const int ct    = tile & 127;               // col tile (2048/16)
    const int khalf = (lane >> 4) * 8;
    const int n     = lane & 15;

    // A row for this lane: row index = t*64 + b
    const int mrow = rt * 16 + n;
    const int t    = mrow >> 6;
    const int b    = mrow & 63;
    const float* arow = emb + (long)seq[b * TT + t] * EE;

    // B row for this lane: W_ih row g (gates = x @ W_ih^T)
    const int g = ct * 16 + n;
    const float* brow = W_ih + (long)g * EE;

    const float bias = b_ih[g] + b_hh[g];
    v8f acc;
    #pragma unroll
    for (int r = 0; r < 8; ++r) acc[r] = bias;

    #pragma unroll
    for (int kk = 0; kk < 10; ++kk) {
        const bool tail = (kk == 9);
        v16bf a = load_frag_f32(arow, kk * 32, khalf, tail);
        v16bf w = load_frag_f32(brow, kk * 32, khalf, tail);
        acc = __builtin_amdgcn_wmma_f32_16x16x32_bf16(
                  false, a, false, w, (short)0, acc, false, false);
    }

    // C/D layout: VGPR r, lane l -> row = rt*16 + r + 8*(l>>4), col = g
    const int rbase = rt * 16 + 8 * (lane >> 4);
    #pragma unroll
    for (int r = 0; r < 8; ++r)
        xg[(long)(rbase + r) * G4 + g] = acc[r];
}

// ---------------------------------------------------------------------------
// Kernel 2: persistent recurrence. 32 blocks x 128 thr.
//   wave (mb = wid 0..3, ut = blockIdx.x 0..31) owns batch rows mb*16..+15
//   and hidden units ut*16..+15 for ALL FOUR gates -> i,f,g,o local in regs.
//   W_hh bf16 tiles cached in LDS once (64 KB/block), reused all 512 steps.
//   h state double-buffered in ws (bf16); one atomic grid barrier per step.
// ---------------------------------------------------------------------------
__global__ void __launch_bounds__(128, 1)
k_rnn(const float* __restrict__ xg, const float* __restrict__ W_hh,
      const int* __restrict__ lengths, __bf16* __restrict__ hbuf,
      unsigned* __restrict__ cnt, float* __restrict__ out) {
    // [gate][kk][lane][16 bf16] = 4*16*32*16*2B = 64 KB
    __shared__ __attribute__((aligned(32))) __bf16 ldsB[4 * 16 * 32 * 16];

    const int lane  = threadIdx.x & 31;
    const int wid   = threadIdx.x >> 5;     // 0..3  (= mb, and = fill gate)
    const int mb    = wid;
    const int ut    = blockIdx.x;           // 0..31
    const int khalf = (lane >> 4) * 8;
    const int n     = lane & 15;
    const int u     = ut * 16 + n;          // hidden unit 0..511

    // ---- One-time fill: wave `wid` converts gate tile `wid` of W_hh to LDS.
    {
        const int grow = wid * HCC + u;     // W_hh row (gates = h @ W_hh^T)
        const float* wrow = W_hh + (long)grow * HCC;
        #pragma unroll
        for (int kk = 0; kk < 16; ++kk) {
            v16bf w = load_frag_f32(wrow, kk * 32, khalf, false);
            *(v16bf*)&ldsB[((wid * 16 + kk) * 32 + lane) * 16] = w;
        }
    }
    __syncthreads();

    // Per-lane persistent state: 8 (batch,unit) cells (C/D layout rows).
    const int rrow0 = mb * 16 + 8 * (lane >> 4);
    int   lens[8];
    float hstate[8], cstate[8];
    #pragma unroll
    for (int r = 0; r < 8; ++r) {
        lens[r]   = lengths[rrow0 + r];
        hstate[r] = 0.0f;
        cstate[r] = 0.0f;
    }

    for (int t = 0; t < TT; ++t) {
        const __bf16* __restrict__ hread  = hbuf + (t & 1) * (BB * HCC);
        __bf16* __restrict__       hwrite = hbuf + ((t + 1) & 1) * (BB * HCC);
        const float* xrow = xg + (long)t * (BB * G4);

        // C init from precomputed input gates (all four gate columns).
        v8f Ci, Cf, Cg, Co;
        #pragma unroll
        for (int r = 0; r < 8; ++r) {
            const float* p = xrow + (long)(rrow0 + r) * G4 + u;
            Ci[r] = p[0];
            Cf[r] = p[HCC];
            Cg[r] = p[2 * HCC];
            Co[r] = p[3 * HCC];
        }
        if (t + 1 < TT) {   // global_prefetch_b8 of next step's xg slice
            __builtin_prefetch(xrow + (long)BB * G4 + (long)rrow0 * G4 + u, 0, 1);
        }

        // gates += h_bf16 @ W_hh^T : K=512 -> 16 bf16-WMMA per gate.
        const __bf16* ha = hread + (mb * 16 + n) * HCC + khalf;
        #pragma unroll
        for (int kk = 0; kk < 16; ++kk) {
            V16 a;
            *(v8bf*)&a.e[0] = *(const v8bf*)(ha + kk * 32);
            *(v8bf*)&a.e[8] = *(const v8bf*)(ha + kk * 32 + 16);
            const v16bf b0 = *(const v16bf*)&ldsB[((0 * 16 + kk) * 32 + lane) * 16];
            const v16bf b1 = *(const v16bf*)&ldsB[((1 * 16 + kk) * 32 + lane) * 16];
            const v16bf b2 = *(const v16bf*)&ldsB[((2 * 16 + kk) * 32 + lane) * 16];
            const v16bf b3 = *(const v16bf*)&ldsB[((3 * 16 + kk) * 32 + lane) * 16];
            Ci = __builtin_amdgcn_wmma_f32_16x16x32_bf16(false, a.v, false, b0, (short)0, Ci, false, false);
            Cf = __builtin_amdgcn_wmma_f32_16x16x32_bf16(false, a.v, false, b1, (short)0, Cf, false, false);
            Cg = __builtin_amdgcn_wmma_f32_16x16x32_bf16(false, a.v, false, b2, (short)0, Cg, false, false);
            Co = __builtin_amdgcn_wmma_f32_16x16x32_bf16(false, a.v, false, b3, (short)0, Co, false, false);
        }

        // Elementwise LSTM cell + variable-length masking, all local.
        #pragma unroll
        for (int r = 0; r < 8; ++r) {
            const int   b_r = rrow0 + r;
            const float ig  = sigmoidf_(Ci[r]);
            const float fg  = sigmoidf_(Cf[r]);
            const float gg  = tanhf(Cg[r]);
            const float og  = sigmoidf_(Co[r]);
            const float cn  = fg * cstate[r] + ig * gg;
            const float hn  = og * tanhf(cn);
            const bool  m   = (t < lens[r]);
            const long  ob  = (long)b_r * (TT * HCC) + (long)t * HCC + u;
            out[ob]                        = m ? hn : 0.0f;   // output
            out[(long)BB * TT * HCC + ob]  = m ? cn : 0.0f;   // cell_output
            if (m) { cstate[r] = cn; hstate[r] = hn; }
            hwrite[b_r * HCC + u] = (__bf16)hstate[r];
        }

        // Grid barrier: release h writes, wait for all 32 blocks.
        __threadfence();
        __syncthreads();
        if (threadIdx.x == 0) {
            atomicAdd(cnt, 1u);
            const unsigned tgt = (unsigned)(t + 1) * NBLK;
            while (atomicAdd(cnt, 0u) < tgt) { __builtin_amdgcn_s_sleep(1); }
        }
        __syncthreads();
        __threadfence();   // acquire: invalidate stale h lines before reuse
    }

    // Final (length-frozen) h and c.
    const long hoff = 2L * BB * TT * HCC;
    #pragma unroll
    for (int r = 0; r < 8; ++r) {
        out[hoff + (long)(rrow0 + r) * HCC + u]            = hstate[r];
        out[hoff + BB * HCC + (long)(rrow0 + r) * HCC + u] = cstate[r];
    }
}

// ---------------------------------------------------------------------------
extern "C" void kernel_launch(void* const* d_in, const int* in_sizes, int n_in,
                              void* d_out, int out_size, void* d_ws, size_t ws_size,
                              hipStream_t stream) {
    const int*   seq     = (const int*)d_in[0];
    const int*   lengths = (const int*)d_in[1];
    const float* emb     = (const float*)d_in[2];
    const float* W_ih    = (const float*)d_in[3];
    const float* W_hh    = (const float*)d_in[4];
    const float* b_ih    = (const float*)d_in[5];
    const float* b_hh    = (const float*)d_in[6];
    float*       out     = (float*)d_out;

    // Workspace layout (needs ~268.6 MB):
    //   xg   f32 [T][B][2048]            : 268,435,456 B
    //   hbuf bf16 [2][B][HC]             :     131,072 B
    //   cnt  unsigned (grid barrier)     :          64 B
    char*     ws   = (char*)d_ws;
    float*    xg   = (float*)ws;
    __bf16*   hbuf = (__bf16*)(ws + (size_t)268435456);
    unsigned* cnt  = (unsigned*)(ws + (size_t)268435456 + 131072);

    // 0) zero h double-buffer + barrier counter
    k_init<<<256, 256, 0, stream>>>(hbuf, cnt);

    // 1) embedding gather + input GEMM (262144 tiles, 8 waves/block)
    k_embed_gemm<<<32768, 256, 0, stream>>>(seq, emb, W_ih, b_ih, b_hh, xg);

    // 2) persistent recurrence: 32 blocks x 128 threads (must co-reside)
    k_rnn<<<NBLK, 128, 0, stream>>>(xg, W_hh, lengths, hbuf, cnt, out);
}
